// TwoIterMultiScaleFusion_63797444214819
// MI455X (gfx1250) — compile-verified, matrix operand order checked by reading, not run on hardware
//
#include <hip/hip_runtime.h>

// ---------------------------------------------------------------------------
// Types for CDNA5 WMMA (wave32, gfx1250)
// ---------------------------------------------------------------------------
typedef __attribute__((ext_vector_type(16))) __bf16 v16bf;
typedef __attribute__((ext_vector_type(8)))  float  v8f;

#define GEMM_WAVES 4   // waves per block
#define GEMM_NT    2   // 16x16 N-tiles per wave (A-fragment reuse)

// ---------------------------------------------------------------------------
// GEMM: Y[M,N] = epilogue( X[M,K](ldx) @ W[N,K]^T + bias ) (+ res)
// Wave computes a 16 x (16*GEMM_NT) strip with v_wmma_f32_16x16x32_bf16.
// act: 0=none, 1=GELU(exact), 2=softplus
// Hot loop is branch-free: K-tail peeled, OOB handled by clamp+select.
// ---------------------------------------------------------------------------
__global__ void gemm_bf16_wmma(const float* __restrict__ X, int ldx,
                               const float* __restrict__ W,
                               const float* __restrict__ bias,
                               const float* __restrict__ res,
                               float* __restrict__ Y, int ldy,
                               int M, int N, int K, int act)
{
    int lane = threadIdx.x & 31;
    int wid  = threadIdx.x >> 5;
    int tm   = blockIdx.y << 4;
    int tn   = (blockIdx.x * GEMM_WAVES + wid) * (16 * GEMM_NT);

    int arow  = tm + (lane & 15);              // A fragment: lane -> M row
    int khalf = (lane & 16) ? 8 : 0;           // K split across half-waves (ISA layout)

    const float* xr = X + (long)arow * ldx + khalf;
    const float* wr[GEMM_NT];
#pragma unroll
    for (int t = 0; t < GEMM_NT; ++t) {
        int brow = tn + 16 * t + (lane & 15);  // B fragment: lane -> N column (row of W)
        int browc = brow < N ? brow : N - 1;   // clamp for safe loads; garbage cols never stored
        wr[t] = W + (long)browc * K + khalf;
    }

    v8f acc[GEMM_NT];
#pragma unroll
    for (int t = 0; t < GEMM_NT; ++t) acc[t] = (v8f){};

    int kfull = K & ~31;
    int k0 = 0;
    for (; k0 < kfull; k0 += 32) {
        v16bf a, b[GEMM_NT];
#pragma unroll
        for (int e = 0; e < 8; ++e) {
            a[e]     = (__bf16)xr[k0 + e];
            a[e + 8] = (__bf16)xr[k0 + 16 + e];
        }
#pragma unroll
        for (int t = 0; t < GEMM_NT; ++t)
#pragma unroll
            for (int e = 0; e < 8; ++e) {
                b[t][e]     = (__bf16)wr[t][k0 + e];
                b[t][e + 8] = (__bf16)wr[t][k0 + 16 + e];
            }
        __builtin_prefetch(xr + k0 + 32, 0, 1);      // gfx1250 global_prefetch_b8
#pragma unroll
        for (int t = 0; t < GEMM_NT; ++t)
            acc[t] = __builtin_amdgcn_wmma_f32_16x16x32_bf16(
                         false, a, false, b[t], (short)0, acc[t], false, false);
    }

    if (k0 < K) {                               // masked K-tail (branch-free selects)
        v16bf a, b[GEMM_NT];
        const float* xb = X + (long)arow * ldx;
#pragma unroll
        for (int e = 0; e < 16; ++e) {
            int kk  = k0 + ((e >> 3) << 4) + khalf + (e & 7);
            int kkc = kk < K ? kk : 0;
            float xa = xb[kkc];
            a[e] = (__bf16)(kk < K ? xa : 0.f);
        }
#pragma unroll
        for (int t = 0; t < GEMM_NT; ++t) {
            const float* wb = wr[t] - khalf;
#pragma unroll
            for (int e = 0; e < 16; ++e) {
                int kk  = k0 + ((e >> 3) << 4) + khalf + (e & 7);
                int kkc = kk < K ? kk : 0;
                float xw = wb[kkc];
                b[t][e] = (__bf16)(kk < K ? xw : 0.f);
            }
        }
#pragma unroll
        for (int t = 0; t < GEMM_NT; ++t)
            acc[t] = __builtin_amdgcn_wmma_f32_16x16x32_bf16(
                         false, a, false, b[t], (short)0, acc[t], false, false);
    }

    int mofs = (lane & 16) ? 8 : 0;
#pragma unroll
    for (int t = 0; t < GEMM_NT; ++t) {
        int col = tn + 16 * t + (lane & 15);
        if (col >= N) continue;
        float bv = bias ? bias[col] : 0.f;
#pragma unroll
        for (int r = 0; r < 8; ++r) {
            int m = tm + r + mofs;
            float v = acc[t][r] + bv;
            if (act == 1)      v = 0.5f * v * (1.f + erff(v * 0.70710678118654752f));
            else if (act == 2) v = fmaxf(v, 0.f) + log1pf(expf(-fabsf(v)));
            if (res) v += res[(long)m * ldy + col];
            Y[(long)m * ldy + col] = v;
        }
    }
}

// ---------------------------------------------------------------------------
// Row-wise LayerNorm (eps 1e-5). Safe in-place. One block per row.
// ---------------------------------------------------------------------------
__global__ void ln_kernel(const float* __restrict__ x,
                          const float* __restrict__ g,
                          const float* __restrict__ b,
                          float* __restrict__ y, int D)
{
    __shared__ float red[256];
    long row = blockIdx.x;
    const float* xr = x + row * D;
    float s = 0.f;
    for (int i = threadIdx.x; i < D; i += blockDim.x) s += xr[i];
    red[threadIdx.x] = s; __syncthreads();
    for (int off = blockDim.x >> 1; off > 0; off >>= 1) {
        if ((int)threadIdx.x < off) red[threadIdx.x] += red[threadIdx.x + off];
        __syncthreads();
    }
    float mean = red[0] / D;
    __syncthreads();
    float v = 0.f;
    for (int i = threadIdx.x; i < D; i += blockDim.x) {
        float t = xr[i] - mean; v += t * t;
    }
    red[threadIdx.x] = v; __syncthreads();
    for (int off = blockDim.x >> 1; off > 0; off >>= 1) {
        if ((int)threadIdx.x < off) red[threadIdx.x] += red[threadIdx.x + off];
        __syncthreads();
    }
    float rstd = rsqrtf(red[0] / D + 1e-5f);
    float* yr = y + row * D;
    for (int i = threadIdx.x; i < D; i += blockDim.x)
        yr[i] = (xr[i] - mean) * rstd * g[i] + b[i];
}

// ---------------------------------------------------------------------------
// Causal depthwise conv1d (kernel 4) + SiLU.  xi = view of xz (stride 2*di).
// ---------------------------------------------------------------------------
__global__ void conv_silu_kernel(const float* __restrict__ xz,
                                 const float* __restrict__ cw,
                                 const float* __restrict__ cb,
                                 float* __restrict__ xc,
                                 int B, int L, int di)
{
    long i = (long)blockIdx.x * blockDim.x + threadIdx.x;
    long total = (long)B * L * di;
    if (i >= total) return;
    int c = (int)(i % di); long r = i / di;
    int l = (int)(r % L);  int bb = (int)(r / L);
    long base = ((long)bb * L) * 2 * di + c;
    float s = cb[c];
#pragma unroll
    for (int k = 0; k < 4; ++k) {
        int ls = l - 3 + k;
        if (ls >= 0) s += xz[base + (long)ls * 2 * di] * cw[c * 4 + k];
    }
    xc[i] = s / (1.f + expf(-s));
}

// ---------------------------------------------------------------------------
// Selective scan: 16 lanes per (b,d) channel, lane n owns state n.
// h = exp(dt*A)*h + dt*B*x; y = sum_n h*C (shfl-xor reduce) then +x*D, *silu(z)
// ---------------------------------------------------------------------------
__global__ void scan_kernel(const float* __restrict__ dt,
                            const float* __restrict__ xc,
                            const float* __restrict__ xdb, int ldxdb, int dr,
                            const float* __restrict__ A_log,
                            const float* __restrict__ Dp,
                            const float* __restrict__ xz,     // z at offset di, stride 2*di
                            float* __restrict__ yout,
                            int B, int L, int di)
{
    long t = (long)blockIdx.x * blockDim.x + threadIdx.x;
    long nch = (long)B * di;
    int n = (int)(t & 15);
    long ch = t >> 4;
    if (ch >= nch) return;
    int d = (int)(ch % di); int bb = (int)(ch / di);
    float A = -expf(A_log[d * 16 + n]);
    float h = 0.f;
    long rowbase = (long)bb * L;
    for (int l = 0; l < L; ++l) {
        long row = rowbase + l;
        float dtv = dt[row * di + d];
        float xcv = xc[row * di + d];
        float Bv  = xdb[row * ldxdb + dr + n];
        float Cv  = xdb[row * ldxdb + dr + 16 + n];
        h = expf(dtv * A) * h + dtv * Bv * xcv;
        float p = h * Cv;
        p += __shfl_xor(p, 1, 16);
        p += __shfl_xor(p, 2, 16);
        p += __shfl_xor(p, 4, 16);
        p += __shfl_xor(p, 8, 16);
        if (n == 0) {
            float z  = xz[row * 2 * di + di + d];
            float sz = z / (1.f + expf(-z));
            yout[row * di + d] = (p + xcv * Dp[d]) * sz;
        }
    }
}

// ---------------------------------------------------------------------------
// Bilinear resize, align_corners=False (NHWC)
// ---------------------------------------------------------------------------
__global__ void resize_kernel(const float* __restrict__ x, float* __restrict__ y,
                              int B, int H, int W, int C, int OH, int OW)
{
    long i = (long)blockIdx.x * blockDim.x + threadIdx.x;
    long total = (long)B * OH * OW * C;
    if (i >= total) return;
    int c = (int)(i % C); long r = i / C;
    int ox = (int)(r % OW); r /= OW;
    int oy = (int)(r % OH); int bb = (int)(r / OH);
    float cy = fminf(fmaxf((oy + 0.5f) * (float)H / OH - 0.5f, 0.f), (float)H - 1.f);
    float cx = fminf(fmaxf((ox + 0.5f) * (float)W / OW - 0.5f, 0.f), (float)W - 1.f);
    int y0 = (int)floorf(cy); int y1 = min(y0 + 1, H - 1); float fy = cy - y0;
    int x0 = (int)floorf(cx); int x1 = min(x0 + 1, W - 1); float fx = cx - x0;
    long base = (long)bb * H * W * C;
    float v00 = x[base + ((long)y0 * W + x0) * C + c];
    float v01 = x[base + ((long)y0 * W + x1) * C + c];
    float v10 = x[base + ((long)y1 * W + x0) * C + c];
    float v11 = x[base + ((long)y1 * W + x1) * C + c];
    float top = v00 * (1.f - fx) + v01 * fx;
    float bot = v10 * (1.f - fx) + v11 * fx;
    y[i] = top * (1.f - fy) + bot * fy;
}

__global__ void add_kernel(const float* __restrict__ a, const float* __restrict__ b,
                           float* __restrict__ y, long n)
{
    long i = (long)blockIdx.x * blockDim.x + threadIdx.x;
    if (i < n) y[i] = a[i] + b[i];
}

__global__ void concat2_kernel(const float* __restrict__ a, const float* __restrict__ b,
                               float* __restrict__ y, long M, int C)
{
    long i = (long)blockIdx.x * blockDim.x + threadIdx.x;
    long total = M * 2 * C;
    if (i >= total) return;
    int col = (int)(i % (2 * C)); long row = i / (2 * C);
    y[i] = (col < C) ? a[row * C + col] : b[row * C + (col - C)];
}

__global__ void bngelu_kernel(float* __restrict__ h,
                              const float* __restrict__ m, const float* __restrict__ v,
                              const float* __restrict__ g, const float* __restrict__ b,
                              long M, int C)
{
    long i = (long)blockIdx.x * blockDim.x + threadIdx.x;
    if (i >= M * (long)C) return;
    int c = (int)(i % C);
    float x = (h[i] - m[c]) * rsqrtf(v[c] + 1e-5f) * g[c] + b[c];
    h[i] = 0.5f * x * (1.f + erff(x * 0.70710678118654752f));
}

__global__ void aaf_mix_kernel(const float* __restrict__ f0, const float* __restrict__ f2,
                               const float* __restrict__ lg, float* __restrict__ y,
                               long M, int C)
{
    long i = (long)blockIdx.x * blockDim.x + threadIdx.x;
    if (i >= M * (long)C) return;
    long row = i / C;
    float l0 = lg[row * 2 + 0], l1 = lg[row * 2 + 1];
    float a0 = 1.f / (1.f + expf((l1 - l0) * 0.5f));   // softmax w/ TEMP=2
    y[i] = f0[i] * a0 + f2[i] * (1.f - a0);
}

// ---------------------------------------------------------------------------
// Host-side orchestration
// ---------------------------------------------------------------------------
struct LNP  { const float *b, *g; };
struct LinP { const float *b, *w; };
struct AAFP { const float *bn_b, *bn_g, *bn_m, *bn_v, *w1, *w2; };
struct BlkP {
    const float *ff_b1, *ff_b2, *ff_w1, *ff_w2;
    const float *ln1_b, *ln1_g, *ln2_b, *ln2_g;
    const float *A_log, *Dp, *conv_b, *conv_w, *dt_b, *dt_w, *in_w, *out_w, *xp_w;
};
struct Params {
    AAFP aaf;
    LinP ad01, ad10, ad12, ad21;
    LNP  ln_c0_c1, ln_c1_c0, ln_c1_c2, ln_c2_c1;
    LNP  src0, src1, src2;
    BlkP s0[5], s1[5], s2[5];
};
struct Ws {
    float *s0, *s1, *s2, *t2, *f0, *f2, *cc, *ah, *lg, *amix, *g0, *rbuf, *stack;
};

static inline void gemm(hipStream_t st, const float* X, int ldx, const float* W,
                        const float* bias, const float* res, float* Y, int ldy,
                        int M, int N, int K, int act)
{
    const int colsPerBlock = 16 * GEMM_NT * GEMM_WAVES;
    dim3 blk(32 * GEMM_WAVES);
    dim3 grd((N + colsPerBlock - 1) / colsPerBlock, M / 16);
    gemm_bf16_wmma<<<grd, blk, 0, st>>>(X, ldx, W, bias, res, Y, ldy, M, N, K, act);
}
static inline void ln(hipStream_t st, const float* x, const float* g, const float* b,
                      float* y, int M, int D)
{
    ln_kernel<<<M, 256, 0, st>>>(x, g, b, y, D);
}
static inline long eb(long n) { return (n + 255) / 256; }

static void run_stack(hipStream_t st, float* x, int B, int L, int d,
                      const BlkP* blks, float* sc)
{
    int di = 2 * d, dr = (d + 15) / 16, h4 = 4 * d, M = B * L;
    float* t   = sc; sc += (size_t)M * d;
    float* xz  = sc; sc += (size_t)M * 2 * di;
    float* xc  = sc; sc += (size_t)M * di;
    float* xdb = sc; sc += (size_t)M * (dr + 32);
    float* dt  = sc; sc += (size_t)M * di;
    float* ys  = sc; sc += (size_t)M * di;
    float* fh  = sc;
    for (int k = 0; k < 5; ++k) {
        const BlkP& p = blks[k];
        ln(st, x, p.ln1_g, p.ln1_b, t, M, d);
        gemm(st, t, d, p.in_w, nullptr, nullptr, xz, 2 * di, M, 2 * di, d, 0);
        conv_silu_kernel<<<eb((long)M * di), 256, 0, st>>>(xz, p.conv_w, p.conv_b, xc, B, L, di);
        gemm(st, xc, di, p.xp_w, nullptr, nullptr, xdb, dr + 32, M, dr + 32, di, 0);
        gemm(st, xdb, dr + 32, p.dt_w, p.dt_b, nullptr, dt, di, M, di, dr, 2);
        scan_kernel<<<eb((long)B * di * 16), 256, 0, st>>>(dt, xc, xdb, dr + 32, dr,
                                                           p.A_log, p.Dp, xz, ys, B, L, di);
        gemm(st, ys, di, p.out_w, nullptr, x, x, d, M, d, di, 0);        // x += mamba(ln1 x)
        ln(st, x, p.ln2_g, p.ln2_b, t, M, d);
        gemm(st, t, d, p.ff_w1, p.ff_b1, nullptr, fh, h4, M, h4, d, 1);  // GELU
        gemm(st, fh, h4, p.ff_w2, p.ff_b2, x, x, d, M, d, h4, 0);        // x += ffn
    }
}

static void fusion_step(hipStream_t st, const Params& P, const Ws& w,
                        const float* c0, const float* c1, const float* c2,
                        float* n0, float* n1, float* n2)
{
    const int B = 2;
    ln(st, c0, P.src0.g, P.src0.b, w.s0, B * 4096, 96);
    ln(st, c1, P.src1.g, P.src1.b, w.s1, B * 1024, 192);
    ln(st, c2, P.src2.g, P.src2.b, w.s2, B * 256, 768);

    // c2 branch: t = ln(adapt(s1, 32x32 -> 16x16))
    resize_kernel<<<eb((long)B * 256 * 192), 256, 0, st>>>(w.s1, w.rbuf, B, 32, 32, 192, 16, 16);
    gemm(st, w.rbuf, 192, P.ad12.w, P.ad12.b, nullptr, w.t2, 768, B * 256, 768, 192, 0);
    ln(st, w.t2, P.ln_c1_c2.g, P.ln_c1_c2.b, w.t2, B * 256, 768);
    run_stack(st, w.t2, B, 256, 768, P.s2, w.stack);
    add_kernel<<<eb((long)B * 256 * 768), 256, 0, st>>>(c2, w.t2, n2, (long)B * 256 * 768);

    // c1 branch: AAF(f0, f2)
    resize_kernel<<<eb((long)B * 1024 * 96), 256, 0, st>>>(w.s0, w.rbuf, B, 64, 64, 96, 32, 32);
    gemm(st, w.rbuf, 96, P.ad01.w, P.ad01.b, nullptr, w.f0, 192, B * 1024, 192, 96, 0);
    ln(st, w.f0, P.ln_c0_c1.g, P.ln_c0_c1.b, w.f0, B * 1024, 192);
    resize_kernel<<<eb((long)B * 1024 * 768), 256, 0, st>>>(w.s2, w.rbuf, B, 16, 16, 768, 32, 32);
    gemm(st, w.rbuf, 768, P.ad21.w, P.ad21.b, nullptr, w.f2, 192, B * 1024, 192, 768, 0);
    ln(st, w.f2, P.ln_c2_c1.g, P.ln_c2_c1.b, w.f2, B * 1024, 192);
    concat2_kernel<<<eb((long)B * 1024 * 384), 256, 0, st>>>(w.f0, w.f2, w.cc, (long)B * 1024, 192);
    gemm(st, w.cc, 384, P.aaf.w1, nullptr, nullptr, w.ah, 1536, B * 1024, 1536, 384, 0);
    bngelu_kernel<<<eb((long)B * 1024 * 1536), 256, 0, st>>>(w.ah, P.aaf.bn_m, P.aaf.bn_v,
                                                             P.aaf.bn_g, P.aaf.bn_b,
                                                             (long)B * 1024, 1536);
    gemm(st, w.ah, 1536, P.aaf.w2, nullptr, nullptr, w.lg, 2, B * 1024, 2, 1536, 0);
    aaf_mix_kernel<<<eb((long)B * 1024 * 192), 256, 0, st>>>(w.f0, w.f2, w.lg, w.amix,
                                                             (long)B * 1024, 192);
    run_stack(st, w.amix, B, 1024, 192, P.s1, w.stack);
    add_kernel<<<eb((long)B * 1024 * 192), 256, 0, st>>>(c1, w.amix, n1, (long)B * 1024 * 192);

    // c0 branch: g = ln(adapt(s1, 32x32 -> 64x64))
    resize_kernel<<<eb((long)B * 4096 * 192), 256, 0, st>>>(w.s1, w.rbuf, B, 32, 32, 192, 64, 64);
    gemm(st, w.rbuf, 192, P.ad10.w, P.ad10.b, nullptr, w.g0, 96, B * 4096, 96, 192, 0);
    ln(st, w.g0, P.ln_c1_c0.g, P.ln_c1_c0.b, w.g0, B * 4096, 96);
    run_stack(st, w.g0, B, 4096, 96, P.s0, w.stack);
    add_kernel<<<eb((long)B * 4096 * 96), 256, 0, st>>>(c0, w.g0, n0, (long)B * 4096 * 96);
}

extern "C" void kernel_launch(void* const* d_in, const int* in_sizes, int n_in,
                              void* d_out, int out_size, void* d_ws, size_t ws_size,
                              hipStream_t stream)
{
    (void)in_sizes;
    if (n_in < 286) return;

    const float* c0_in = (const float*)d_in[0];
    const float* c1_in = (const float*)d_in[1];
    const float* c2_in = (const float*)d_in[2];

    // ---- walk params in JAX sorted-key pytree order ----
    int idx = 3;
    auto nxt = [&]() { return (const float*)d_in[idx++]; };
    Params P;
    P.aaf.bn_b = nxt(); P.aaf.bn_g = nxt(); P.aaf.bn_m = nxt(); P.aaf.bn_v = nxt();
    P.aaf.w1 = nxt();   P.aaf.w2 = nxt();
    P.ad01.b = nxt(); P.ad01.w = nxt();
    P.ad10.b = nxt(); P.ad10.w = nxt();
    P.ad12.b = nxt(); P.ad12.w = nxt();
    P.ad21.b = nxt(); P.ad21.w = nxt();
    P.ln_c0_c1.b = nxt(); P.ln_c0_c1.g = nxt();
    P.ln_c1_c0.b = nxt(); P.ln_c1_c0.g = nxt();
    P.ln_c1_c2.b = nxt(); P.ln_c1_c2.g = nxt();
    P.ln_c2_c1.b = nxt(); P.ln_c2_c1.g = nxt();
    P.src0.b = nxt(); P.src0.g = nxt();
    P.src1.b = nxt(); P.src1.g = nxt();
    P.src2.b = nxt(); P.src2.g = nxt();
    BlkP* stacks[3] = { P.s0, P.s1, P.s2 };
    for (int s = 0; s < 3; ++s)
        for (int k = 0; k < 5; ++k) {
            BlkP& bp = stacks[s][k];
            bp.ff_b1 = nxt(); bp.ff_b2 = nxt(); bp.ff_w1 = nxt(); bp.ff_w2 = nxt();
            bp.ln1_b = nxt(); bp.ln1_g = nxt(); bp.ln2_b = nxt(); bp.ln2_g = nxt();
            bp.A_log = nxt(); bp.Dp = nxt();
            bp.conv_b = nxt(); bp.conv_w = nxt();
            bp.dt_b = nxt();  bp.dt_w = nxt();
            bp.in_w = nxt();  bp.out_w = nxt(); bp.xp_w = nxt();
        }

    // ---- workspace bump allocator ----
    const size_t S0 = 2l * 4096 * 96, S1 = 2l * 1024 * 192, S2 = 2l * 256 * 768;
    float* ws = (float*)d_ws;
    size_t off = 0;
    auto alloc = [&](size_t n) { float* p = ws + off; off += n; return p; };

    float* A0 = alloc(S0); float* A1 = alloc(S1); float* A2 = alloc(S2);
    float* B0 = alloc(S0); float* B1 = alloc(S1); float* B2 = alloc(S2);
    Ws w;
    w.s0 = alloc(S0); w.s1 = alloc(S1); w.s2 = alloc(S2);
    w.t2 = alloc(S2);
    w.f0 = alloc(S1); w.f2 = alloc(S1);
    w.cc = alloc(2l * 1024 * 384);
    w.ah = alloc(2l * 1024 * 1536);
    w.lg = alloc(2l * 1024 * 2);
    w.amix = alloc(S1);
    w.g0   = alloc(S0);
    w.rbuf = alloc(2l * 4096 * 192);        // max resize output
    // stack scratch: worst case is the c0 stack (M=8192, d=96) -> ~12.2M floats
    size_t stack_floats = 0;
    {
        int Bc = 2;
        int Ls[3] = {4096, 1024, 256}, ds[3] = {96, 192, 768};
        for (int i = 0; i < 3; ++i) {
            int d = ds[i], di = 2 * d, dr = (d + 15) / 16;
            size_t M = (size_t)Bc * Ls[i];
            size_t need = M * d + M * 2 * di + M * di + M * (dr + 32)
                        + M * di + M * di + M * 4 * d;
            if (need > stack_floats) stack_floats = need;
        }
    }
    w.stack = alloc(stack_floats);
    if (off * sizeof(float) > ws_size) return;   // insufficient scratch: bail safely

    // ---- copy initial state (do not mutate inputs) ----
    hipMemcpyAsync(A0, c0_in, S0 * sizeof(float), hipMemcpyDeviceToDevice, stream);
    hipMemcpyAsync(A1, c1_in, S1 * sizeof(float), hipMemcpyDeviceToDevice, stream);
    hipMemcpyAsync(A2, c2_in, S2 * sizeof(float), hipMemcpyDeviceToDevice, stream);

    // ---- two fusion iterations (ping-pong) ----
    fusion_step(stream, P, w, A0, A1, A2, B0, B1, B2);
    fusion_step(stream, P, w, B0, B1, B2, A0, A1, A2);

    // ---- pack outputs (c0n, c1n, c2n) ----
    float* out = (float*)d_out;
    hipMemcpyAsync(out,            A0, S0 * sizeof(float), hipMemcpyDeviceToDevice, stream);
    hipMemcpyAsync(out + S0,       A1, S1 * sizeof(float), hipMemcpyDeviceToDevice, stream);
    hipMemcpyAsync(out + S0 + S1,  A2, S2 * sizeof(float), hipMemcpyDeviceToDevice, stream);
    (void)out_size;
}
